// GraphiTLayer_67912022884407
// MI455X (gfx1250) — matrix operand order repaired and checked by reading, not compiled, not run on hardware
//
#include <hip/hip_runtime.h>
#include <hip/hip_bf16.h>
#include <math.h>

#define NN 4096
#define DD 128
#define HH 8
#define EE 131072
#define SCALE 0.25f

typedef __attribute__((ext_vector_type(16))) __bf16 v16bf;
typedef __attribute__((ext_vector_type(8)))  __bf16 v8bf;
typedef __attribute__((ext_vector_type(8)))  float  v8f;

// ---- WMMA fragment loaders (CDNA5 16x16x32 bf16 layouts, wave32) ----
// A (16xK tile, row-major source): lane holds row = lane&15; elements 0..7 are
// K = k0+8*half .. +7, elements 8..15 are K = k0+16+8*half .. +7.
static __device__ __forceinline__ v16bf fragA(const __bf16* row, int k0, int half) {
  v8bf a = *(const v8bf*)(row + k0 + 8 * half);
  v8bf b = *(const v8bf*)(row + k0 + 16 + 8 * half);
  v16bf f;
#pragma unroll
  for (int i = 0; i < 8; ++i) { f[i] = a[i]; f[i + 8] = b[i]; }
  return f;
}
// B (KxN tile = W^T, W row-major [Nout,K]): lane holds col = lane&15; elements
// 0..7 are K = k0+8*half .. +7, elements 8..15 are K = k0+16+8*half .. +7.
static __device__ __forceinline__ v16bf fragB(const __bf16* row, int k0, int half) {
  v8bf a = *(const v8bf*)(row + k0 + 8 * half);
  v8bf b = *(const v8bf*)(row + k0 + 16 + 8 * half);
  v16bf f;
#pragma unroll
  for (int i = 0; i < 8; ++i) { f[i] = a[i]; f[i + 8] = b[i]; }
  return f;
}

// ---- generic WMMA GEMM: C[M,Nout] (+)= A[M,K] @ W[Nout,K]^T (+ bias) ----
__global__ void __launch_bounds__(256)
k_gemm(const __bf16* __restrict__ A, int lda,
       const __bf16* __restrict__ W, int ldw,
       float* __restrict__ C, int ldc,
       const float* __restrict__ bias,
       int M, int Nout, int K, int accum) {
  int lane = threadIdx.x & 31;
  int wv = threadIdx.x >> 5;
  int ntn = Nout >> 4;
  int ntiles = (M >> 4) * ntn;
  int tile = blockIdx.x * (blockDim.x >> 5) + wv;
  if (tile >= ntiles) return;  // wave-uniform
  int tm = tile / ntn, tn = tile - tm * ntn;
  int l15 = lane & 15, half = lane >> 4;
  const __bf16* arow = A + (size_t)((tm << 4) + l15) * lda;
  const __bf16* wrow = W + (size_t)((tn << 4) + l15) * ldw;
  v8f acc;
#pragma unroll
  for (int r = 0; r < 8; ++r) acc[r] = 0.0f;
  for (int k0 = 0; k0 < K; k0 += 32) {
    v16bf a = fragA(arow, k0, half);
    v16bf b = fragB(wrow, k0, half);
    acc = __builtin_amdgcn_wmma_f32_16x16x32_bf16(false, a, false, b, (short)0, acc,
                                                  false, false);
  }
  int col = (tn << 4) + l15;
  float bv = bias ? bias[col] : 0.0f;
#pragma unroll
  for (int r = 0; r < 8; ++r) {
    int m = (tm << 4) + r + 8 * half;
    float v = acc[r] + bv;
    float* p = C + (size_t)m * ldc + col;
    if (accum) *p += v; else *p = v;
  }
}

// ---- flash attention: one wave = one (head, 16-query tile); keys in 32-chunks ----
__global__ void __launch_bounds__(256)
k_flash(const __bf16* __restrict__ Qh, const __bf16* __restrict__ Kh,
        const __bf16* __restrict__ VT, __bf16* __restrict__ Oh) {
  __shared__ __bf16 lds[8][16 * 32];  // per-wave 16x32 P staging tile
  int lane = threadIdx.x & 31;
  int wv = threadIdx.x >> 5;
  int gw = blockIdx.x * 8 + wv;       // 0..2047
  int h = gw >> 8;
  int q0 = (gw & 255) << 4;
  int l15 = lane & 15, half = lane >> 4;
  __bf16* myl = &lds[wv][0];

  // Q fragment, head_dim 16 padded to K=32 with zeros; SCALE folded in.
  v16bf qf;
  {
    const __bf16* qp = Qh + (size_t)(q0 + l15) * DD + h * 16 + 8 * half;
    v8bf c = *(const v8bf*)qp;
#pragma unroll
    for (int i = 0; i < 8; ++i) {
      qf[i] = (__bf16)((float)c[i] * SCALE);
      qf[i + 8] = (__bf16)0.0f;
    }
  }
  float m_i[8], l_i[8];
  v8f O;
#pragma unroll
  for (int r = 0; r < 8; ++r) { m_i[r] = -3.0e38f; l_i[r] = 0.0f; O[r] = 0.0f; }

  for (int key0 = 0; key0 < NN; key0 += 32) {
    // K^T B-fragments: contraction K = head_dim (16); elements 0..7 hold d = 8*half+i, 8..15 are zero pad.
    v16bf kf0, kf1;
#pragma unroll
    for (int i = 0; i < 16; ++i) { kf0[i] = (__bf16)0.0f; kf1[i] = (__bf16)0.0f; }
    {
      const __bf16* kp0 = Kh + (size_t)(key0 + l15) * DD + h * 16 + 8 * half;
      const __bf16* kp1 = Kh + (size_t)(key0 + 16 + l15) * DD + h * 16 + 8 * half;
      v8bf a0 = *(const v8bf*)kp0;
      v8bf a1 = *(const v8bf*)kp1;
#pragma unroll
      for (int i = 0; i < 8; ++i) {
        kf0[i] = a0[i];
        kf1[i] = a1[i];
      }
    }
    v8f z;
#pragma unroll
    for (int r = 0; r < 8; ++r) z[r] = 0.0f;
    v8f S0 = __builtin_amdgcn_wmma_f32_16x16x32_bf16(false, qf, false, kf0, (short)0, z,
                                                     false, false);
    v8f S1 = __builtin_amdgcn_wmma_f32_16x16x32_bf16(false, qf, false, kf1, (short)0, z,
                                                     false, false);
    // online softmax; row r lives in lanes of same half at col = lane&15
#pragma unroll
    for (int r = 0; r < 8; ++r) {
      float mx = fmaxf(S0[r], S1[r]);
#pragma unroll
      for (int d = 8; d >= 1; d >>= 1) mx = fmaxf(mx, __shfl_xor(mx, d, 16));
      float mn = fmaxf(m_i[r], mx);
      float corr = __expf(m_i[r] - mn);
      float p0 = __expf(S0[r] - mn);
      float p1 = __expf(S1[r] - mn);
      float sm = p0 + p1;
#pragma unroll
      for (int d = 8; d >= 1; d >>= 1) sm += __shfl_xor(sm, d, 16);
      l_i[r] = l_i[r] * corr + sm;
      m_i[r] = mn;
      O[r] *= corr;
      int mrow = r + 8 * half;
      myl[mrow * 32 + l15] = (__bf16)p0;
      myl[mrow * 32 + 16 + l15] = (__bf16)p1;
    }
    asm volatile("s_wait_dscnt 0" ::: "memory");  // P staged -> safe to re-read
    v16bf pf, vf;
    {
      const __bf16* pr = myl + l15 * 32;
      v8bf a = *(const v8bf*)(pr + 8 * half);
      v8bf b = *(const v8bf*)(pr + 16 + 8 * half);
#pragma unroll
      for (int i = 0; i < 8; ++i) { pf[i] = a[i]; pf[i + 8] = b[i]; }
    }
    {
      const __bf16* vr = VT + (size_t)(h * 16 + l15) * NN + key0 + 8 * half;
      v8bf a = *(const v8bf*)vr;
      v8bf b = *(const v8bf*)(vr + 16);
#pragma unroll
      for (int i = 0; i < 8; ++i) { vf[i] = a[i]; vf[i + 8] = b[i]; }
    }
    O = __builtin_amdgcn_wmma_f32_16x16x32_bf16(false, pf, false, vf, (short)0, O,
                                                false, false);
  }
#pragma unroll
  for (int r = 0; r < 8; ++r) {
    float v = O[r] / l_i[r];
    Oh[(size_t)(q0 + r + 8 * half) * DD + h * 16 + l15] = (__bf16)v;
  }
}

// ---- edge attention helpers ----
static __device__ __forceinline__ unsigned fkey(float f) {
  unsigned u = __float_as_uint(f);
  return (u & 0x80000000u) ? ~u : (u | 0x80000000u);
}
static __device__ __forceinline__ float funkey(unsigned k) {
  unsigned u = (k & 0x80000000u) ? (k & 0x7FFFFFFFu) : ~k;
  return __uint_as_float(u);
}

__global__ void k_edge_scores(const long long* __restrict__ ei,
                              const float* __restrict__ Q, const float* __restrict__ K,
                              float* __restrict__ attn, unsigned* __restrict__ mkey) {
  int e = blockIdx.x * blockDim.x + threadIdx.x;
  if (e >= EE) return;
  int src = (int)ei[e];
  int dst = (int)ei[EE + e];
  const float* q = Q + (size_t)dst * DD;
  const float* k = K + (size_t)src * DD;
#pragma unroll
  for (int h = 0; h < HH; ++h) {
    float s = 0.0f;
#pragma unroll
    for (int d = 0; d < 16; ++d) s += q[h * 16 + d] * k[h * 16 + d];
    s *= SCALE;
    attn[(size_t)e * HH + h] = s;
    atomicMax(&mkey[dst * HH + h], fkey(s));
  }
}

__global__ void k_edge_accum(const long long* __restrict__ ei,
                             const float* __restrict__ V, const float* __restrict__ attn,
                             const unsigned* __restrict__ mkey,
                             float* __restrict__ ssum, float* __restrict__ aggU) {
  int e = blockIdx.x * blockDim.x + threadIdx.x;
  if (e >= EE) return;
  int src = (int)ei[e];
  int dst = (int)ei[EE + e];
#pragma unroll
  for (int h = 0; h < HH; ++h) {
    float m = funkey(mkey[dst * HH + h]);
    float ev = __expf(attn[(size_t)e * HH + h] - m);
    atomicAdd(&ssum[dst * HH + h], ev);
    const float* vs = V + (size_t)src * DD + h * 16;
    float* ag = aggU + (size_t)dst * DD + h * 16;
#pragma unroll
    for (int d = 0; d < 16; ++d) atomicAdd(&ag[d], vs[d] * ev);
  }
}

__global__ void k_norm_agg(float* __restrict__ aggU, const float* __restrict__ ssum,
                           __bf16* __restrict__ aggh) {
  int i = blockIdx.x * blockDim.x + threadIdx.x;
  if (i >= NN * DD) return;
  int n = i >> 7, c = i & 127, h = c >> 4;
  float sv = ssum[n * HH + h];
  float a = (sv > 0.0f) ? aggU[i] / sv : 0.0f;
  aggU[i] = a;
  aggh[i] = (__bf16)a;
}

__global__ void k_local(const float* __restrict__ x, const float* __restrict__ agg,
                        const float* __restrict__ gatepre, __bf16* __restrict__ localh) {
  int i = blockIdx.x * blockDim.x + threadIdx.x;
  if (i >= NN * DD) return;
  float g = 1.0f / (1.0f + __expf(-gatepre[i]));
  localh[i] = (__bf16)(g * agg[i] + (1.0f - g) * x[i]);
}

static __device__ __forceinline__ float wsum(float v) {
#pragma unroll
  for (int d = 16; d >= 1; d >>= 1) v += __shfl_xor(v, d, 32);
  return v;
}

// LayerNorm (wave per row of 128) -> bf16 output
__global__ void k_ln_bf(const float* __restrict__ X, const float* __restrict__ g,
                        const float* __restrict__ b, __bf16* __restrict__ outh) {
  int row = (blockIdx.x * blockDim.x + threadIdx.x) >> 5;
  int lane = threadIdx.x & 31;
  if (row >= NN) return;
  const float* x = X + (size_t)row * DD;
  float v[4];
#pragma unroll
  for (int i = 0; i < 4; ++i) v[i] = x[lane + 32 * i];
  float mean = wsum(v[0] + v[1] + v[2] + v[3]) * (1.0f / DD);
  float var = 0.0f;
#pragma unroll
  for (int i = 0; i < 4; ++i) { float d = v[i] - mean; var += d * d; }
  var = wsum(var) * (1.0f / DD);
  float rs = rsqrtf(var + 1e-5f);
#pragma unroll
  for (int i = 0; i < 4; ++i) {
    int c = lane + 32 * i;
    outh[(size_t)row * DD + c] = (__bf16)((v[i] - mean) * rs * g[c] + b[c]);
  }
}

// fused = relu(ln(fusedpre,gf,betaf)); h = x + fused -> hout; h2 = ln(h,g2,be2) -> bf16
__global__ void k_fuse(const float* __restrict__ fp_, const float* __restrict__ x,
                       const float* __restrict__ gf, const float* __restrict__ bfv,
                       const float* __restrict__ g2, const float* __restrict__ be2,
                       float* __restrict__ hout, __bf16* __restrict__ h2h) {
  int row = (blockIdx.x * blockDim.x + threadIdx.x) >> 5;
  int lane = threadIdx.x & 31;
  if (row >= NN) return;
  const float* fr = fp_ + (size_t)row * DD;
  const float* xr = x + (size_t)row * DD;
  float v[4];
#pragma unroll
  for (int i = 0; i < 4; ++i) v[i] = fr[lane + 32 * i];
  float mean = wsum(v[0] + v[1] + v[2] + v[3]) * (1.0f / DD);
  float var = 0.0f;
#pragma unroll
  for (int i = 0; i < 4; ++i) { float d = v[i] - mean; var += d * d; }
  var = wsum(var) * (1.0f / DD);
  float rs = rsqrtf(var + 1e-5f);
  float hr[4];
#pragma unroll
  for (int i = 0; i < 4; ++i) {
    int c = lane + 32 * i;
    float y = fmaxf((v[i] - mean) * rs * gf[c] + bfv[c], 0.0f);
    hr[i] = xr[c] + y;
    hout[(size_t)row * DD + c] = hr[i];
  }
  float mean2 = wsum(hr[0] + hr[1] + hr[2] + hr[3]) * (1.0f / DD);
  float var2 = 0.0f;
#pragma unroll
  for (int i = 0; i < 4; ++i) { float d = hr[i] - mean2; var2 += d * d; }
  var2 = wsum(var2) * (1.0f / DD);
  float rs2 = rsqrtf(var2 + 1e-5f);
#pragma unroll
  for (int i = 0; i < 4; ++i) {
    int c = lane + 32 * i;
    h2h[(size_t)row * DD + c] = (__bf16)((hr[i] - mean2) * rs2 * g2[c] + be2[c]);
  }
}

__global__ void k_gelu(const float* __restrict__ mid, __bf16* __restrict__ midh) {
  int i = blockIdx.x * blockDim.x + threadIdx.x;
  if (i >= NN * 4 * DD) return;
  float xv = mid[i];
  midh[i] = (__bf16)(0.5f * xv * (1.0f + erff(xv * 0.70710678118f)));
}

__global__ void k_f2bf(const float* __restrict__ s, __bf16* __restrict__ d, int n) {
  int i = blockIdx.x * blockDim.x + threadIdx.x;
  if (i < n) d[i] = (__bf16)s[i];
}

// Vg [N,D] fp32 -> VgT [D,N] bf16
__global__ void k_tcvt(const float* __restrict__ s, __bf16* __restrict__ d) {
  int i = blockIdx.x * blockDim.x + threadIdx.x;
  if (i >= NN * DD) return;
  int n = i >> 7, c = i & 127;
  d[(size_t)c * NN + n] = (__bf16)s[i];
}

extern "C" void kernel_launch(void* const* d_in, const int* in_sizes, int n_in,
                              void* d_out, int out_size, void* d_ws, size_t ws_size,
                              hipStream_t stream) {
  (void)in_sizes; (void)n_in; (void)out_size; (void)ws_size;
  const float* x = (const float*)d_in[0];
  const long long* ei = (const long long*)d_in[1];
  const float* wq_l = (const float*)d_in[2];  const float* bq_l = (const float*)d_in[3];
  const float* wk_l = (const float*)d_in[4];  const float* bk_l = (const float*)d_in[5];
  const float* wv_l = (const float*)d_in[6];  const float* bv_l = (const float*)d_in[7];
  const float* wg   = (const float*)d_in[8];  const float* bg   = (const float*)d_in[9];
  const float* wq_g = (const float*)d_in[10]; const float* bq_g = (const float*)d_in[11];
  const float* wk_g = (const float*)d_in[12]; const float* bk_g = (const float*)d_in[13];
  const float* wv_g = (const float*)d_in[14]; const float* bv_g = (const float*)d_in[15];
  const float* wo_g = (const float*)d_in[16]; const float* bo_g = (const float*)d_in[17];
  const float* wf   = (const float*)d_in[18]; const float* bfv  = (const float*)d_in[19];
  const float* gf   = (const float*)d_in[20]; const float* betaf= (const float*)d_in[21];
  const float* w1   = (const float*)d_in[22]; const float* b1   = (const float*)d_in[23];
  const float* w2   = (const float*)d_in[24]; const float* b2   = (const float*)d_in[25];
  const float* g1   = (const float*)d_in[26]; const float* be1  = (const float*)d_in[27];
  const float* g2   = (const float*)d_in[28]; const float* be2  = (const float*)d_in[29];

  char* base = (char*)d_ws;
  size_t off = 0;
  auto alloc = [&](size_t bytes) -> void* {
    off = (off + 255) & ~(size_t)255;
    void* p = base + off;
    off += bytes;
    return p;
  };
  __bf16* xh    = (__bf16*)alloc((size_t)NN * DD * 2);
  __bf16* hwq_l = (__bf16*)alloc((size_t)DD * DD * 2);
  __bf16* hwk_l = (__bf16*)alloc((size_t)DD * DD * 2);
  __bf16* hwv_l = (__bf16*)alloc((size_t)DD * DD * 2);
  __bf16* hwg   = (__bf16*)alloc((size_t)DD * 2 * DD * 2);
  __bf16* hwq_g = (__bf16*)alloc((size_t)DD * DD * 2);
  __bf16* hwk_g = (__bf16*)alloc((size_t)DD * DD * 2);
  __bf16* hwv_g = (__bf16*)alloc((size_t)DD * DD * 2);
  __bf16* hwo_g = (__bf16*)alloc((size_t)DD * DD * 2);
  __bf16* hwf   = (__bf16*)alloc((size_t)DD * 2 * DD * 2);
  __bf16* hw1   = (__bf16*)alloc((size_t)4 * DD * DD * 2);
  __bf16* hw2   = (__bf16*)alloc((size_t)DD * 4 * DD * 2);
  float*  Qf    = (float*)alloc((size_t)NN * DD * 4);   // Ql then Qg (sequential reuse)
  float*  Kf    = (float*)alloc((size_t)NN * DD * 4);
  float*  Vf    = (float*)alloc((size_t)NN * DD * 4);
  float*  attn  = (float*)alloc((size_t)EE * HH * 4);
  unsigned* mkey= (unsigned*)alloc((size_t)NN * HH * 4);
  float*  ssum  = (float*)alloc((size_t)NN * HH * 4);
  float*  aggU  = (float*)alloc((size_t)NN * DD * 4);
  __bf16* aggh  = (__bf16*)alloc((size_t)NN * DD * 2);
  float*  gatep = (float*)alloc((size_t)NN * DD * 4);
  __bf16* localh= (__bf16*)alloc((size_t)NN * DD * 2);
  __bf16* xnh   = (__bf16*)alloc((size_t)NN * DD * 2);
  __bf16* Qgh   = (__bf16*)alloc((size_t)NN * DD * 2);
  __bf16* Kgh   = (__bf16*)alloc((size_t)NN * DD * 2);
  __bf16* VgT   = (__bf16*)alloc((size_t)DD * NN * 2);
  __bf16* aoh   = (__bf16*)alloc((size_t)NN * DD * 2);
  float*  go    = (float*)alloc((size_t)NN * DD * 4);
  __bf16* goh   = (__bf16*)alloc((size_t)NN * DD * 2);
  float*  fpre  = (float*)alloc((size_t)NN * DD * 4);
  __bf16* h2h   = (__bf16*)alloc((size_t)NN * DD * 2);
  float*  mid   = (float*)alloc((size_t)NN * 4 * DD * 4);
  __bf16* midh  = (__bf16*)alloc((size_t)NN * 4 * DD * 2);

  auto cvt = [&](const float* s, __bf16* d, int n) {
    k_f2bf<<<(n + 255) / 256, 256, 0, stream>>>(s, d, n);
  };
  auto gemm = [&](const __bf16* A, int lda, const __bf16* W, int ldw, float* C, int ldc,
                  const float* bias, int M, int Nout, int K, int accum) {
    int tiles = (M / 16) * (Nout / 16);
    k_gemm<<<(tiles + 7) / 8, 256, 0, stream>>>(A, lda, W, ldw, C, ldc, bias, M, Nout, K,
                                                accum);
  };

  hipMemsetAsync(mkey, 0, (size_t)NN * HH * 4, stream);
  hipMemsetAsync(ssum, 0, (size_t)NN * HH * 4, stream);
  hipMemsetAsync(aggU, 0, (size_t)NN * DD * 4, stream);

  cvt(x, xh, NN * DD);
  cvt(wq_l, hwq_l, DD * DD); cvt(wk_l, hwk_l, DD * DD); cvt(wv_l, hwv_l, DD * DD);
  cvt(wg, hwg, DD * 2 * DD);
  cvt(wq_g, hwq_g, DD * DD); cvt(wk_g, hwk_g, DD * DD); cvt(wv_g, hwv_g, DD * DD);
  cvt(wo_g, hwo_g, DD * DD);
  cvt(wf, hwf, DD * 2 * DD);
  cvt(w1, hw1, 4 * DD * DD); cvt(w2, hw2, DD * 4 * DD);

  // ---- local branch ----
  gemm(xh, DD, hwq_l, DD, Qf, DD, bq_l, NN, DD, DD, 0);
  gemm(xh, DD, hwk_l, DD, Kf, DD, bk_l, NN, DD, DD, 0);
  gemm(xh, DD, hwv_l, DD, Vf, DD, bv_l, NN, DD, DD, 0);
  k_edge_scores<<<EE / 256, 256, 0, stream>>>(ei, Qf, Kf, attn, mkey);
  k_edge_accum<<<EE / 256, 256, 0, stream>>>(ei, Vf, attn, mkey, ssum, aggU);
  k_norm_agg<<<(NN * DD) / 256, 256, 0, stream>>>(aggU, ssum, aggh);
  gemm(xh, DD, hwg, 2 * DD, gatep, DD, bg, NN, DD, DD, 0);          // wg[:, :128]
  gemm(aggh, DD, hwg + DD, 2 * DD, gatep, DD, nullptr, NN, DD, DD, 1);  // wg[:, 128:]
  k_local<<<(NN * DD) / 256, 256, 0, stream>>>(x, aggU, gatep, localh);

  // ---- global dense MHA ----
  k_ln_bf<<<NN / 8, 256, 0, stream>>>(x, g1, be1, xnh);
  gemm(xnh, DD, hwq_g, DD, Qf, DD, bq_g, NN, DD, DD, 0);
  gemm(xnh, DD, hwk_g, DD, Kf, DD, bk_g, NN, DD, DD, 0);
  gemm(xnh, DD, hwv_g, DD, Vf, DD, bv_g, NN, DD, DD, 0);
  cvt(Qf, Qgh, NN * DD);
  cvt(Kf, Kgh, NN * DD);
  k_tcvt<<<(NN * DD) / 256, 256, 0, stream>>>(Vf, VgT);
  k_flash<<<(HH * (NN / 16)) / 8, 256, 0, stream>>>(Qgh, Kgh, VgT, aoh);
  gemm(aoh, DD, hwo_g, DD, go, DD, bo_g, NN, DD, DD, 0);
  cvt(go, goh, NN * DD);

  // ---- fusion + residual + FFN ----
  gemm(localh, DD, hwf, 2 * DD, fpre, DD, bfv, NN, DD, DD, 0);        // wf[:, :128]
  gemm(goh, DD, hwf + DD, 2 * DD, fpre, DD, nullptr, NN, DD, DD, 1);  // wf[:, 128:]
  k_fuse<<<NN / 8, 256, 0, stream>>>(fpre, x, gf, betaf, g2, be2, (float*)d_out, h2h);
  gemm(h2h, DD, hw1, DD, mid, 4 * DD, b1, NN, 4 * DD, DD, 0);
  k_gelu<<<(NN * 4 * DD) / 256, 256, 0, stream>>>(mid, midh);
  gemm(midh, 4 * DD, hw2, 4 * DD, (float*)d_out, DD, b2, NN, DD, 4 * DD, 1);
}
